// MultiQueryAttentionBatched_12781822673473
// MI455X (gfx1250) — matrix operand order repaired and checked by reading, not compiled, not run on hardware
//
#include <hip/hip_runtime.h>

// Problem dims (fixed by reference)
#define BB   2
#define NN   2048
#define MM   2048
#define DD   1024
#define HH   16
#define KK   64
#define VV   64

typedef __attribute__((ext_vector_type(16))) __bf16 v16bf;
typedef __attribute__((ext_vector_type(8)))  __bf16 v8bf;
typedef __attribute__((ext_vector_type(8)))  float  v8f;

static __device__ __forceinline__ __bf16 f2bf(float f) {
  unsigned u = __builtin_bit_cast(unsigned, f);
  unsigned r = (u + 0x7FFFu + ((u >> 16) & 1u)) >> 16;   // RNE
  unsigned short s = (unsigned short)r;
  return __builtin_bit_cast(__bf16, s);
}

static __device__ __forceinline__ v16bf cat16(v8bf lo, v8bf hi) {
  return __builtin_shufflevector(lo, hi, 0,1,2,3,4,5,6,7,8,9,10,11,12,13,14,15);
}

static __device__ __forceinline__ v8f wmma_bf16(v16bf a, v16bf b, v8f c) {
  return __builtin_amdgcn_wmma_f32_16x16x32_bf16(false, a, false, b, (short)0, c, false, false);
}

// One 16-byte async global->LDS copy (CDNA5 GV mode, tracked by ASYNCcnt).
static __device__ __forceinline__ void async_cp16(unsigned lds_off,
                                                  unsigned long long gaddr) {
  asm volatile("global_load_async_to_lds_b128 %0, %1, off"
               :: "v"(lds_off), "v"(gaddr) : "memory");
}
static __device__ __forceinline__ void wait_async0() {
  asm volatile("s_wait_asynccnt 0" ::: "memory");
}

// ---------------- conversion / layout kernels ----------------

__global__ void k_cvt_bf16(const float* __restrict__ src, __bf16* __restrict__ dst,
                           long long count) {
  long long i = (long long)blockIdx.x * blockDim.x + threadIdx.x;
  if (i < count) dst[i] = f2bf(src[i]);
}

// PqT[h][k][d] = P_q[h][d][k]
__global__ void k_cvt_pq(const float* __restrict__ src, __bf16* __restrict__ dst) {
  long long i = (long long)blockIdx.x * blockDim.x + threadIdx.x;
  int kk = (int)(i % KK);
  int dd = (int)((i / KK) % DD);
  int hh = (int)(i / (KK * DD));
  dst[((long long)hh * KK + kk) * DD + dd] = f2bf(src[i]);
}

// PkT[k][d] = P[d][k]   (also used for P_v)
__global__ void k_cvt_pk(const float* __restrict__ src, __bf16* __restrict__ dst) {
  long long i = (long long)blockIdx.x * blockDim.x + threadIdx.x;
  int kk = (int)(i % KK);
  int dd = (int)(i / KK);
  dst[(long long)kk * DD + dd] = f2bf(src[i]);
}

// PoT[d][h*VV+v] = P_o[h][d][v]   (weight stored [N=d, K=h*v])
__global__ void k_cvt_po(const float* __restrict__ src, __bf16* __restrict__ dst) {
  long long i = (long long)blockIdx.x * blockDim.x + threadIdx.x;
  int vv = (int)(i % VV);
  int dd = (int)((i / VV) % DD);
  int hh = (int)(i / (VV * DD));
  dst[(long long)dd * (HH * VV) + hh * VV + vv] = f2bf(src[i]);
}

// ---------------- generic bf16 WMMA GEMM ----------------
// C[m,n] = sum_k A[m,k] * Bt[n,k];   A row-major [M,K], Bt row-major [N,K].
// 256 threads = 8 waves, one 16x16 tile per wave; block tile 64x32.
// mode 0: f32 row-major; 1: bf16 row-major; 2: bf16 transposed ([N,M]).
__global__ void k_gemm_bf16(const __bf16* __restrict__ A, long long lda,
                            long long strideAz, int zDivA,
                            const __bf16* __restrict__ Bt, long long ldb,
                            long long strideBz, int zModB,
                            void* __restrict__ Cout, long long ldc,
                            long long strideCz, int Kdim, int mode) {
  const int lane = threadIdx.x & 31;
  const int wv   = threadIdx.x >> 5;
  const int wm   = wv & 3;
  const int wn   = wv >> 2;
  const int m0   = blockIdx.x * 64 + wm * 16;
  const int n0   = blockIdx.y * 32 + wn * 16;
  const int z    = blockIdx.z;
  const int hl   = lane >> 4;
  const int l16  = lane & 15;
  const int kbA  = hl * 16;
  const int kbB  = hl * 16;

  const __bf16* a  = A  + (long long)(z / zDivA) * strideAz
                        + (long long)(m0 + l16) * lda;
  const __bf16* bt = Bt + (long long)(z % zModB) * strideBz
                        + (long long)(n0 + l16) * ldb;

  v8f acc = {};
  for (int kk = 0; kk < Kdim; kk += 32) {
    v16bf av = cat16(*(const v8bf*)(a + kk + kbA),
                     *(const v8bf*)(a + kk + 8 + kbA));
    v16bf bv = *(const v16bf*)(bt + kk + kbB);
    acc = wmma_bf16(av, bv, acc);
  }

  if (mode == 0) {
    float* C = (float*)Cout + (long long)z * strideCz;
#pragma unroll
    for (int r = 0; r < 8; ++r)
      C[(long long)(m0 + r + hl * 8) * ldc + n0 + l16] = acc[r];
  } else if (mode == 1) {
    __bf16* C = (__bf16*)Cout + (long long)z * strideCz;
#pragma unroll
    for (int r = 0; r < 8; ++r)
      C[(long long)(m0 + r + hl * 8) * ldc + n0 + l16] = f2bf(acc[r]);
  } else {
    __bf16* C = (__bf16*)Cout + (long long)z * strideCz;
#pragma unroll
    for (int r = 0; r < 8; ++r)
      C[(long long)(n0 + l16) * ldc + (m0 + r + hl * 8)] = f2bf(acc[r]);
  }
}

// ---------------- flash attention ----------------
// Qb [b,h,n,KK] bf16, Kb [b,m,KK] bf16, VTb [b,VV,m] bf16, mask [b,h,n,m] f32,
// Ob [b,n,h,VV] bf16.  128 threads = 4 waves; wave owns 16 query rows;
// block sweeps m in 64-key chunks, K/V^T chunks double-buffered in LDS via
// global_load_async_to_lds_b128 (ASYNCcnt).
__global__ void k_attn(const __bf16* __restrict__ Qb, const __bf16* __restrict__ Kb,
                       const __bf16* __restrict__ VTb, const float* __restrict__ mask,
                       __bf16* __restrict__ Ob) {
  __shared__ __align__(64) __bf16 kbuf[2][64 * KK];   // [m-local][k]   8KB each
  __shared__ __align__(64) __bf16 vbuf[2][VV * 64];   // [v][m-local]   8KB each
  __shared__ __align__(64) __bf16 plds[4][16][64];    // per-wave P     8KB total

  const int tid  = threadIdx.x;
  const int lane = tid & 31;
  const int wv   = tid >> 5;
  const int hl   = lane >> 4;
  const int l16  = lane & 15;
  const int kbA  = hl * 16;
  const int kbB  = hl * 16;

  const int ntiles = NN / 64;
  const int nb = blockIdx.x % ntiles;
  const int bh = blockIdx.x / ntiles;        // b*HH + h
  const int b  = bh >> 4;
  const int h  = bh & 15;
  const int n0 = nb * 64 + wv * 16;

  const __bf16* kbase = Kb  + (long long)b * MM * KK;
  const __bf16* vbase = VTb + (long long)b * VV * MM;
  const float*  mrow  = mask + ((long long)bh * NN + n0) * MM;

  // Q fragments (16 rows x 64 K): two 16x32 A tiles, resident all sweep.
  const __bf16* qrow = Qb + ((long long)bh * NN + n0 + l16) * KK;
  v16bf qa0 = cat16(*(const v8bf*)(qrow + 0  + kbA), *(const v8bf*)(qrow + 8 + kbA));
  v16bf qa1 = cat16(*(const v8bf*)(qrow + 32 + kbA), *(const v8bf*)(qrow + 40 + kbA));

  v8f o0 = {}, o1 = {}, o2 = {}, o3 = {};
  float mi[8], li[8];
#pragma unroll
  for (int r = 0; r < 8; ++r) { mi[r] = -1e30f; li[r] = 0.0f; }

  // cooperative async stage of one 64-key K + V^T chunk into LDS buffer bi
  auto stage = [&](int mmc, int bi) {
    // K chunk: 64 rows x 128B, contiguous 8KB
    const char* gk = (const char*)(kbase + (long long)mmc * KK);
    char* lk = (char*)&kbuf[bi][0];
#pragma unroll
    for (int s = 0; s < 4; ++s) {
      unsigned o = (unsigned)((s * 128 + tid) * 16);
      async_cp16((unsigned)(size_t)(lk + o), (unsigned long long)(size_t)(gk + o));
    }
    // V^T chunk: 64 v-rows x 128B pieces, row stride MM*2 bytes
    const int vr = tid >> 3;
    const int vc = (tid & 7) * 16;
    const char* gv = (const char*)vbase + (long long)mmc * 2 + vc;
    char* lv = (char*)&vbuf[bi][0] + vc;
#pragma unroll
    for (int s = 0; s < 4; ++s) {
      int row = s * 16 + vr;
      async_cp16((unsigned)(size_t)(lv + row * 128),
                 (unsigned long long)(size_t)(gv + (long long)row * (MM * 2)));
    }
  };

  stage(0, 0);

  for (int mm = 0; mm < MM; mm += 64) {
    const int cur = (mm >> 6) & 1;
    wait_async0();        // our chunk's async copies done
    __syncthreads();      // everyone's copies done; prev buffer fully consumed
    if (mm + 64 < MM) stage(mm + 64, cur ^ 1);

    // ---- S = Q K^T for 4 key tiles (ds_load B fragments) ----
    const __bf16* kc = &kbuf[cur][0];
    v8f s0 = {}, s1 = {}, s2 = {}, s3 = {};
    s0 = wmma_bf16(qa0, *(const v16bf*)(kc + ( 0 + l16) * KK + 0  + kbB), s0);
    s0 = wmma_bf16(qa1, *(const v16bf*)(kc + ( 0 + l16) * KK + 32 + kbB), s0);
    s1 = wmma_bf16(qa0, *(const v16bf*)(kc + (16 + l16) * KK + 0  + kbB), s1);
    s1 = wmma_bf16(qa1, *(const v16bf*)(kc + (16 + l16) * KK + 32 + kbB), s1);
    s2 = wmma_bf16(qa0, *(const v16bf*)(kc + (32 + l16) * KK + 0  + kbB), s2);
    s2 = wmma_bf16(qa1, *(const v16bf*)(kc + (32 + l16) * KK + 32 + kbB), s2);
    s3 = wmma_bf16(qa0, *(const v16bf*)(kc + (48 + l16) * KK + 0  + kbB), s3);
    s3 = wmma_bf16(qa1, *(const v16bf*)(kc + (48 + l16) * KK + 32 + kbB), s3);

    // ---- online softmax over 64 keys; l_i kept as per-lane partial ----
#pragma unroll
    for (int r = 0; r < 8; ++r) {
      const int row = r + hl * 8;
      const float* mp = mrow + (long long)row * MM + mm;
      float e0 = s0[r] + mp[ 0 + l16];
      float e1 = s1[r] + mp[16 + l16];
      float e2 = s2[r] + mp[32 + l16];
      float e3 = s3[r] + mp[48 + l16];
      if (mm + 64 < MM) __builtin_prefetch(mp + 64 + l16, 0, 1);
      float t = fmaxf(fmaxf(e0, e1), fmaxf(e2, e3));
      t = fmaxf(t, __shfl_xor(t, 1, 32));
      t = fmaxf(t, __shfl_xor(t, 2, 32));
      t = fmaxf(t, __shfl_xor(t, 4, 32));
      t = fmaxf(t, __shfl_xor(t, 8, 32));
      float mn = fmaxf(mi[r], t);
      float sc = __expf(mi[r] - mn);
      mi[r] = mn;
      float q0 = __expf(e0 - mn);
      float q1 = __expf(e1 - mn);
      float q2 = __expf(e2 - mn);
      float q3 = __expf(e3 - mn);
      li[r] = li[r] * sc + (q0 + q1 + q2 + q3);
      o0[r] *= sc; o1[r] *= sc; o2[r] *= sc; o3[r] *= sc;
      // C-layout -> LDS (per-wave region; same-wave LDS ops are in-order)
      plds[wv][row][ 0 + l16] = f2bf(q0);
      plds[wv][row][16 + l16] = f2bf(q1);
      plds[wv][row][32 + l16] = f2bf(q2);
      plds[wv][row][48 + l16] = f2bf(q3);
    }

    // P as two 16x32 A fragments
    v16bf pa0 = cat16(*(const v8bf*)&plds[wv][l16][ 0 + kbA],
                      *(const v8bf*)&plds[wv][l16][ 8 + kbA]);
    v16bf pa1 = cat16(*(const v8bf*)&plds[wv][l16][32 + kbA],
                      *(const v8bf*)&plds[wv][l16][40 + kbA]);

    // ---- O += P V : four v-tiles x two 32-key halves ----
    const __bf16* vc = &vbuf[cur][0];
    o0 = wmma_bf16(pa0, *(const v16bf*)(vc + ( 0 + l16) * 64 + 0  + kbB), o0);
    o0 = wmma_bf16(pa1, *(const v16bf*)(vc + ( 0 + l16) * 64 + 32 + kbB), o0);
    o1 = wmma_bf16(pa0, *(const v16bf*)(vc + (16 + l16) * 64 + 0  + kbB), o1);
    o1 = wmma_bf16(pa1, *(const v16bf*)(vc + (16 + l16) * 64 + 32 + kbB), o1);
    o2 = wmma_bf16(pa0, *(const v16bf*)(vc + (32 + l16) * 64 + 0  + kbB), o2);
    o2 = wmma_bf16(pa1, *(const v16bf*)(vc + (32 + l16) * 64 + 32 + kbB), o2);
    o3 = wmma_bf16(pa0, *(const v16bf*)(vc + (48 + l16) * 64 + 0  + kbB), o3);
    o3 = wmma_bf16(pa1, *(const v16bf*)(vc + (48 + l16) * 64 + 32 + kbB), o3);
  }

  // deferred l_i reduction (uniform per-row scale made partials consistent)
#pragma unroll
  for (int r = 0; r < 8; ++r) {
    float s = li[r];
    s += __shfl_xor(s, 1, 32);
    s += __shfl_xor(s, 2, 32);
    s += __shfl_xor(s, 4, 32);
    s += __shfl_xor(s, 8, 32);
    li[r] = s;
  }

  // normalize and store O as [b, n, h, v]
#pragma unroll
  for (int r = 0; r < 8; ++r) {
    float inv = 1.0f / li[r];
    long long nrow = n0 + r + hl * 8;
    __bf16* op = Ob + (((long long)b * NN + nrow) * HH + h) * VV;
    op[ 0 + l16] = f2bf(o0[r] * inv);
    op[16 + l16] = f2bf(o1[r] * inv);
    op[32 + l16] = f2bf(o2[r] * inv);
    op[48 + l16] = f2bf(o3[r] * inv);
  }
}

// ---------------- host launch ----------------

extern "C" void kernel_launch(void* const* d_in, const int* in_sizes, int n_in,
                              void* d_out, int out_size, void* d_ws, size_t ws_size,
                              hipStream_t stream) {
  const float* X    = (const float*)d_in[0];
  const float* M    = (const float*)d_in[1];
  const float* mask = (const float*)d_in[2];
  const float* Pq   = (const float*)d_in[3];
  const float* Pk   = (const float*)d_in[4];
  const float* Pv   = (const float*)d_in[5];
  const float* Po   = (const float*)d_in[6];
  float* Y = (float*)d_out;

  char* ws = (char*)d_ws;
  size_t off = 0;
  auto alloc = [&](size_t bytes) -> void* {
    void* p = ws + off;
    off += (bytes + 255) & ~(size_t)255;
    return p;
  };
  __bf16* Xb  = (__bf16*)alloc((size_t)BB * NN * DD * 2);
  __bf16* Mb  = (__bf16*)alloc((size_t)BB * MM * DD * 2);
  __bf16* PqT = (__bf16*)alloc((size_t)HH * KK * DD * 2);
  __bf16* PkT = (__bf16*)alloc((size_t)KK * DD * 2);
  __bf16* PvT = (__bf16*)alloc((size_t)VV * DD * 2);
  __bf16* PoT = (__bf16*)alloc((size_t)DD * HH * VV * 2);
  __bf16* Qb  = (__bf16*)alloc((size_t)BB * HH * NN * KK * 2);
  __bf16* Kb  = (__bf16*)alloc((size_t)BB * MM * KK * 2);
  __bf16* VTb = (__bf16*)alloc((size_t)BB * VV * MM * 2);
  __bf16* Ob  = (__bf16*)alloc((size_t)BB * NN * HH * VV * 2);

  // ---- precision conversion + weight re-layout ----
  k_cvt_bf16<<<(BB * NN * DD) / 256, 256, 0, stream>>>(X, Xb, (long long)BB * NN * DD);
  k_cvt_bf16<<<(BB * MM * DD) / 256, 256, 0, stream>>>(M, Mb, (long long)BB * MM * DD);
  k_cvt_pq  <<<(HH * DD * KK) / 256, 256, 0, stream>>>(Pq, PqT);
  k_cvt_pk  <<<(DD * KK) / 256,      256, 0, stream>>>(Pk, PkT);
  k_cvt_pk  <<<(DD * VV) / 256,      256, 0, stream>>>(Pv, PvT);
  k_cvt_po  <<<(HH * DD * VV) / 256, 256, 0, stream>>>(Po, PoT);

  // ---- Q = X @ P_q  (per b,h) -> Qb [b,h,n,k] ----
  k_gemm_bf16<<<dim3(NN / 64, KK / 32, BB * HH), 256, 0, stream>>>(
      Xb, DD, (long long)NN * DD, HH,
      PqT, DD, (long long)KK * DD, HH,
      Qb, KK, (long long)NN * KK, DD, 1);

  // ---- K = M @ P_k  (per b) -> Kb [b,m,k] ----
  k_gemm_bf16<<<dim3(MM / 64, KK / 32, BB), 256, 0, stream>>>(
      Mb, DD, (long long)MM * DD, 1,
      PkT, DD, 0, 1,
      Kb, KK, (long long)MM * KK, DD, 1);

  // ---- V = M @ P_v  (per b), stored transposed -> VTb [b,v,m] ----
  k_gemm_bf16<<<dim3(MM / 64, VV / 32, BB), 256, 0, stream>>>(
      Mb, DD, (long long)MM * DD, 1,
      PvT, DD, 0, 1,
      VTb, MM, (long long)VV * MM, DD, 2);

  // ---- flash attention -> Ob [b,n,h,v] ----
  k_attn<<<dim3(BB * HH * (NN / 64)), 128, 0, stream>>>(Qb, Kb, VTb, mask, Ob);

  // ---- Y = O @ P_o^T : [n, h*v] x [h*v, d] per b, f32 out ----
  k_gemm_bf16<<<dim3(NN / 64, DD / 32, BB), 256, 0, stream>>>(
      Ob, HH * VV, (long long)NN * HH * VV, 1,
      PoT, HH * VV, 0, 1,
      Y, DD, (long long)NN * DD, DD, 0);
}
